// INT8QuantizedAttention_6983616824473
// MI455X (gfx1250) — matrix operand order repaired and checked by reading, not compiled, not run on hardware
//
#include <hip/hip_runtime.h>
#include <hip/hip_bf16.h>

typedef _Float16 h16;
typedef __attribute__((ext_vector_type(16))) _Float16 v16h;
typedef __attribute__((ext_vector_type(8)))  _Float16 v8h;
typedef __attribute__((ext_vector_type(8)))  float    v8f;

// ds_swizzle_b32 xor-shuffle within 32 lanes; xor masks < 16 stay inside the
// 16-lane row group (group-of-32 mode: and=0x1F, or=0, xor=m -> imm 0x1F|(m<<10))
#define SWZ_XOR(x, imm) \
  __int_as_float(__builtin_amdgcn_ds_swizzle(__float_as_int(x), (imm)))

// ---------------------------------------------------------------------------
// WMMA helper: D(16x16,f32) = A(16x32,f16) x B(32x16,f16) + C
// ---------------------------------------------------------------------------
__device__ __forceinline__ v8f wmma_f16(v16h a, v16h b, v8f c) {
  return __builtin_amdgcn_wmma_f32_16x16x32_f16(
      /*neg_a=*/false, a, /*neg_b=*/false, b,
      /*c_mod=*/(short)0, c, /*reuse_a=*/false, /*reuse_b=*/false);
}

// A-fragment (16x32 f16). `p` must already point at
// row (lane&15) with the +hh*16 half-lane offset folded in; k0 = K-tile base.
// elements 0..7  -> K = k0 + hh*16 + e       (contiguous 16B)
// elements 8..15 -> K = k0 + hh*16 + e       (contiguous 16B)
__device__ __forceinline__ v16h load_a_at(const h16* __restrict__ p, int k0) {
  v8h lo = *(const v8h*)(p + k0);
  v8h hi = *(const v8h*)(p + k0 + 8);
  v16h a;
#pragma unroll
  for (int e = 0; e < 8; ++e) { a[e] = lo[e]; a[e + 8] = hi[e]; }
  return a;
}

// ---------------------------------------------------------------------------
// f32 -> f16 conversion
// ---------------------------------------------------------------------------
__global__ void f32_to_f16_kernel(const float* __restrict__ in, h16* __restrict__ out,
                                  size_t n) {
  size_t i = (size_t)blockIdx.x * blockDim.x + threadIdx.x;
  if (i < n) out[i] = (h16)in[i];
}

// ---------------------------------------------------------------------------
// WMMA GEMM: C[m,n] = sum_k A[m,k] * Bw[n,k]   (torch Linear: x @ W^T)
// A row-major [M,K] f16, Bw row-major [N,K] f16.
// One wave computes a 32x64 strip (2 A-frags x 4 B-frags = 8 WMMAs per k-step).
// Two-stage ping/pong pipeline (no register rotation movs).
// blockDim = 128 (4 waves). Requires M%32==0, N%64==0, K%64==0.
// ---------------------------------------------------------------------------
template <typename OutT>
__global__ void __launch_bounds__(128)
wmma_gemm_kernel(const h16* __restrict__ A, const h16* __restrict__ Bw,
                 OutT* __restrict__ C, int M, int N, int K) {
  int wave = blockIdx.x * 4 + (threadIdx.x >> 5);
  int nGroups = N >> 6;
  int m0 = (wave / nGroups) << 5;   // 32 output rows
  int n0 = (wave % nGroups) << 6;   // 64 output cols

  int lane = threadIdx.x & 31;
  int ln = lane & 15, hh = lane >> 4;

  // Per-lane base pointers; inner loop only adds 32-bit constants.
  const h16* pa0 = A + (size_t)(m0 + ln) * (size_t)K + hh * 16;  // A rows m0..m0+15
  const h16* pa1 = pa0 + (size_t)16 * (size_t)K;                 // A rows m0+16..m0+31
  const h16* pb  = Bw + (size_t)(n0 + ln) * (size_t)K + hh * 16; // B rows n0+ln (+j*16)
  const size_t bRow = (size_t)16 * (size_t)K;                    // 16 weight rows

  v8f acc[2][4];
#pragma unroll
  for (int t = 0; t < 2; ++t)
#pragma unroll
    for (int j = 0; j < 4; ++j)
#pragma unroll
      for (int r = 0; r < 8; ++r) acc[t][j][r] = 0.f;

  v16h a0A, a0B, b0[4];   // ping buffer
  v16h a1A, a1B, b1[4];   // pong buffer

  // prologue: tile 0 -> ping
  a0A = load_a_at(pa0, 0);
  a0B = load_a_at(pa1, 0);
#pragma unroll
  for (int j = 0; j < 4; ++j) b0[j] = *(const v16h*)(pb + (size_t)j * bRow);

  const int T = K >> 5;   // number of 32-wide k tiles (even)
  for (int t = 0; t < T - 2; t += 2) {
    const int k1 = (t + 1) << 5;
    const int k2 = (t + 2) << 5;
    // stage A: fetch tile t+1 -> pong, consume ping (tile t)
    a1A = load_a_at(pa0, k1);
    a1B = load_a_at(pa1, k1);
#pragma unroll
    for (int j = 0; j < 4; ++j) b1[j] = *(const v16h*)(pb + (size_t)j * bRow + k1);
    __builtin_prefetch(pb + k2 + 64, 0, 1);    // global_prefetch_b8
    __builtin_prefetch(pa0 + k2 + 64, 0, 1);
#pragma unroll
    for (int j = 0; j < 4; ++j) {
      acc[0][j] = wmma_f16(a0A, b0[j], acc[0][j]);
      acc[1][j] = wmma_f16(a0B, b0[j], acc[1][j]);
    }
    // stage B: fetch tile t+2 -> ping, consume pong (tile t+1)
    a0A = load_a_at(pa0, k2);
    a0B = load_a_at(pa1, k2);
#pragma unroll
    for (int j = 0; j < 4; ++j) b0[j] = *(const v16h*)(pb + (size_t)j * bRow + k2);
#pragma unroll
    for (int j = 0; j < 4; ++j) {
      acc[0][j] = wmma_f16(a1A, b1[j], acc[0][j]);
      acc[1][j] = wmma_f16(a1B, b1[j], acc[1][j]);
    }
  }
  // epilogue: ping holds tile T-2; fetch tile T-1 -> pong, consume both
  {
    const int kl = (T - 1) << 5;
    a1A = load_a_at(pa0, kl);
    a1B = load_a_at(pa1, kl);
#pragma unroll
    for (int j = 0; j < 4; ++j) b1[j] = *(const v16h*)(pb + (size_t)j * bRow + kl);
#pragma unroll
    for (int j = 0; j < 4; ++j) {
      acc[0][j] = wmma_f16(a0A, b0[j], acc[0][j]);
      acc[1][j] = wmma_f16(a0B, b0[j], acc[1][j]);
    }
#pragma unroll
    for (int j = 0; j < 4; ++j) {
      acc[0][j] = wmma_f16(a1A, b1[j], acc[0][j]);
      acc[1][j] = wmma_f16(a1B, b1[j], acc[1][j]);
    }
  }

  // C/D layout: lanes 0-15 hold M=r (VGPR r), lanes 16-31 hold M=r+8; N=lane&15.
  OutT* pc = C + (size_t)m0 * (size_t)N + n0;
#pragma unroll
  for (int t = 0; t < 2; ++t)
#pragma unroll
    for (int j = 0; j < 4; ++j)
#pragma unroll
      for (int r = 0; r < 8; ++r)
        pc[(size_t)(t * 16 + r + 8 * hh) * (size_t)N + (size_t)(j * 16 + ln)] =
            (OutT)acc[t][j][r];
}

// ---------------------------------------------------------------------------
// RoPE in place on f16 [B, S, heads*64].  Pair (j, j+32) rotated by t*inv_freq[j].
// ---------------------------------------------------------------------------
__global__ void rope_kernel(h16* __restrict__ X, const int* __restrict__ pos,
                            int S, int heads) {
  size_t idx = (size_t)blockIdx.x * blockDim.x + threadIdx.x;
  int j = (int)(idx & 31);
  size_t grp = idx >> 5;                    // ((b*S + s)*heads + hd)
  int s = (int)((grp / (size_t)heads) % (size_t)S);
  float t = (float)pos[s];
  float invf = __expf(-(float)j * 0.2878231366f);  // ln(10000)/32
  float sn, cs;
  __sincosf(t * invf, &sn, &cs);
  h16* p = X + grp * 64;
  float x1 = (float)p[j];
  float x2 = (float)p[j + 32];
  p[j]      = (h16)(x1 * cs - x2 * sn);
  p[j + 32] = (h16)(x2 * cs + x1 * sn);
}

// ---------------------------------------------------------------------------
// V transpose: [B, S, KVH*64] -> [B, KVH, 64, S]  (so PV B-operand is contiguous)
// ---------------------------------------------------------------------------
__global__ void transpose_v_kernel(const h16* __restrict__ V, h16* __restrict__ Vt,
                                   int S, int KVHn) {
  size_t idx = (size_t)blockIdx.x * blockDim.x + threadIdx.x;
  int d = (int)(idx & 63);
  size_t t = idx >> 6;
  int kv = (int)(t % (size_t)KVHn); t /= (size_t)KVHn;
  int s = (int)(t % (size_t)S);
  int b = (int)(t / (size_t)S);
  Vt[(((size_t)b * KVHn + kv) * 64 + d) * (size_t)S + s] = V[idx];
}

// ---------------------------------------------------------------------------
// Flash attention, one wave (32 threads) per 16-row Q tile of one (b, head).
// Q,K f16 row-major ([B,S,H*64] / [B,S,KVH*64]); Vt f16 [B,KVH,64,S]; mask f32.
// 64-key tiles: 8 score WMMAs + 8 PV WMMAs per tile; one softmax butterfly
// pass per 64 keys (ds_swizzle xor reductions within the 16-lane row group).
// K fragments are streamed (2 live) to stay within the register file.
// ---------------------------------------------------------------------------
__global__ void __launch_bounds__(32)
attn_kernel(const h16* __restrict__ Q, const h16* __restrict__ Kb,
            const h16* __restrict__ Vt, const float* __restrict__ mask,
            h16* __restrict__ O, int Bn, int S, int Hn, int KVHn) {
  (void)Bn;
  const int D = 64;
  __shared__ __align__(32) h16 Plds[16 * 64];

  int qTiles = S >> 4;
  int qt = blockIdx.x % qTiles;
  int hd = (blockIdx.x / qTiles) % Hn;
  int b  = blockIdx.x / (qTiles * Hn);
  int kvh = hd / (Hn / KVHn);

  int lane = threadIdx.x & 31;
  int cn = lane & 15, hh = lane >> 4;
  const int ldq = Hn * D;     // 2048
  const int ldk = KVHn * D;   // 512

  // Q A-fragments (held in registers for the whole key loop)
  const h16* pq = Q + (size_t)(b * S + qt * 16 + cn) * (size_t)ldq + hd * D + hh * 16;
  v16h qf0 = load_a_at(pq, 0);
  v16h qf1 = load_a_at(pq, 32);

  // Per-lane walking pointers; advanced by 64 keys per iteration.
  const h16* pk = Kb + ((size_t)(b * S) + cn) * (size_t)ldk + kvh * D + hh * 16;
  const h16* pv = Vt + ((size_t)(b * KVHn + kvh) * D + cn) * (size_t)S + hh * 16;
  const float* pm = mask + (size_t)b * S * S + (size_t)(qt * 16) * S + cn;
  const size_t kRow16 = (size_t)16 * (size_t)ldk;  // 16 key rows in K
  const size_t vCol16 = (size_t)16 * (size_t)S;    // 16 d-columns in Vt

  float mrow[8], lrow[8];
  v8f acc[4];
#pragma unroll
  for (int r = 0; r < 8; ++r) { mrow[r] = -3.0e38f; lrow[r] = 0.f; }
#pragma unroll
  for (int j = 0; j < 4; ++j)
#pragma unroll
    for (int r = 0; r < 8; ++r) acc[j][r] = 0.f;

  for (int k0 = 0; k0 < S; k0 += 64) {
    // ---- scores: S(16x64) = Q(16x64) . K^T(64x64), four 16x16 column chunks.
    // K fragments streamed per chunk: only 2 live v16h at a time.
    v8f s[4];
#pragma unroll
    for (int c = 0; c < 4; ++c)
#pragma unroll
      for (int r = 0; r < 8; ++r) s[c][r] = 0.f;
    __builtin_prefetch(pk + 4 * kRow16, 0, 1);   // next 64-key tile
#pragma unroll
    for (int c = 0; c < 4; ++c) {
      v16h kbl = *(const v16h*)(pk + (size_t)c * kRow16);
      v16h kbh = *(const v16h*)(pk + (size_t)c * kRow16 + 32);
      s[c] = wmma_f16(qf0, kbl, s[c]);
      s[c] = wmma_f16(qf1, kbh, s[c]);
    }
    // ---- scale + mask + online softmax (row r+8*hh lives in one half-wave)
#pragma unroll
    for (int r = 0; r < 8; ++r) {
      int row = r + 8 * hh;
      float sc[4];
#pragma unroll
      for (int c = 0; c < 4; ++c)
        sc[c] = s[c][r] * 0.125f + pm[(size_t)row * S + k0 + c * 16];
      float mx = fmaxf(fmaxf(sc[0], sc[1]), fmaxf(sc[2], sc[3]));
      mx = fmaxf(mx, SWZ_XOR(mx, 0x041F));
      mx = fmaxf(mx, SWZ_XOR(mx, 0x081F));
      mx = fmaxf(mx, SWZ_XOR(mx, 0x101F));
      mx = fmaxf(mx, SWZ_XOR(mx, 0x201F));
      float mnew = fmaxf(mrow[r], mx);
      float corr = __expf(mrow[r] - mnew);
      float p[4];
      float rs = 0.f;
#pragma unroll
      for (int c = 0; c < 4; ++c) { p[c] = __expf(sc[c] - mnew); rs += p[c]; }
      rs += SWZ_XOR(rs, 0x041F);
      rs += SWZ_XOR(rs, 0x081F);
      rs += SWZ_XOR(rs, 0x101F);
      rs += SWZ_XOR(rs, 0x201F);
      lrow[r] = lrow[r] * corr + rs;
      mrow[r] = mnew;
#pragma unroll
      for (int j = 0; j < 4; ++j) acc[j][r] *= corr;
      // C-layout -> LDS (row-major 16x64) for re-read in A-layout
#pragma unroll
      for (int c = 0; c < 4; ++c) Plds[row * 64 + c * 16 + cn] = (h16)p[c];
    }
    __syncthreads();
    // ---- PV: acc(16x64) += P(16x64) . V(64x64), four 16x16 N chunks, K=64
    v16h pf0 = load_a_at(Plds + cn * 64 + hh * 16, 0);   // keys k0..k0+31
    v16h pf1 = load_a_at(Plds + cn * 64 + hh * 16, 32);  // keys k0+32..k0+63
#pragma unroll
    for (int j = 0; j < 4; ++j) {
      v16h vb0 = *(const v16h*)(pv + (size_t)j * vCol16);
      v16h vb1 = *(const v16h*)(pv + (size_t)j * vCol16 + 32);
      acc[j] = wmma_f16(pf0, vb0, acc[j]);
      acc[j] = wmma_f16(pf1, vb1, acc[j]);
    }
    __syncthreads();
    pk += 4 * kRow16;  // +64 key rows
    pv += 64;          // +64 key columns
  }

  // ---- normalize and store [B,S,H*64] f16
  h16* po = O + (size_t)(b * S + qt * 16) * (size_t)ldq + hd * D;
#pragma unroll
  for (int r = 0; r < 8; ++r) {
    int row = r + 8 * hh;
    float inv = 1.f / lrow[r];
#pragma unroll
    for (int j = 0; j < 4; ++j)
      po[(size_t)row * (size_t)ldq + (size_t)(j * 16 + cn)] = (h16)(acc[j][r] * inv);
  }
}

// ---------------------------------------------------------------------------
// Host-side orchestration
// ---------------------------------------------------------------------------
extern "C" void kernel_launch(void* const* d_in, const int* in_sizes, int n_in,
                              void* d_out, int out_size, void* d_ws, size_t ws_size,
                              hipStream_t stream) {
  (void)in_sizes; (void)n_in; (void)out_size; (void)ws_size;
  const int B = 2, S = 2048, E = 2048, H = 32, KVH = 8, D = 64;

  const float* hidden = (const float*)d_in[0];
  const float* mask   = (const float*)d_in[1];
  const float* Wq     = (const float*)d_in[2];
  const float* Wk     = (const float*)d_in[3];
  const float* Wv     = (const float*)d_in[4];
  const float* Wo     = (const float*)d_in[5];
  const int*   pos    = (const int*)d_in[6];
  float* out = (float*)d_out;

  h16* ws = (h16*)d_ws;
  size_t o = 0;
  h16* hidden16 = ws + o; o += (size_t)B * S * E;        // 16.8 MB
  h16* wq16 = ws + o;     o += (size_t)H * D * E;        //  8.4 MB
  h16* wk16 = ws + o;     o += (size_t)KVH * D * E;      //  2.1 MB
  h16* wv16 = ws + o;     o += (size_t)KVH * D * E;      //  2.1 MB
  h16* wo16 = ws + o;     o += (size_t)E * H * D;        //  8.4 MB
  h16* q16  = ws + o;     o += (size_t)B * S * H * D;    // 16.8 MB
  h16* k16  = ws + o;     o += (size_t)B * S * KVH * D;  //  4.2 MB
  h16* v16  = ws + o;     o += (size_t)B * S * KVH * D;  //  4.2 MB
  h16* vt16 = ws + o;     o += (size_t)B * S * KVH * D;  //  4.2 MB
  h16* ao16 = ws + o;     o += (size_t)B * S * H * D;    // 16.8 MB

  // 1) down-convert operands to f16
  {
    size_t n;
    n = (size_t)B * S * E;
    f32_to_f16_kernel<<<(unsigned)((n + 255) / 256), 256, 0, stream>>>(hidden, hidden16, n);
    n = (size_t)H * D * E;
    f32_to_f16_kernel<<<(unsigned)((n + 255) / 256), 256, 0, stream>>>(Wq, wq16, n);
    n = (size_t)KVH * D * E;
    f32_to_f16_kernel<<<(unsigned)((n + 255) / 256), 256, 0, stream>>>(Wk, wk16, n);
    n = (size_t)KVH * D * E;
    f32_to_f16_kernel<<<(unsigned)((n + 255) / 256), 256, 0, stream>>>(Wv, wv16, n);
    n = (size_t)E * H * D;
    f32_to_f16_kernel<<<(unsigned)((n + 255) / 256), 256, 0, stream>>>(Wo, wo16, n);
  }

  const int M = B * S;  // 4096
  // 2) QKV projections (WMMA GEMMs, 32x64 strip per wave)
  {
    int N = H * D, K = E;
    int blocks = ((M / 32) * (N / 64)) / 4;
    wmma_gemm_kernel<h16><<<blocks, 128, 0, stream>>>(hidden16, wq16, q16, M, N, K);
  }
  {
    int N = KVH * D, K = E;
    int blocks = ((M / 32) * (N / 64)) / 4;
    wmma_gemm_kernel<h16><<<blocks, 128, 0, stream>>>(hidden16, wk16, k16, M, N, K);
    wmma_gemm_kernel<h16><<<blocks, 128, 0, stream>>>(hidden16, wv16, v16, M, N, K);
  }

  // 3) RoPE + V transpose
  {
    size_t n = (size_t)B * S * H * 32;
    rope_kernel<<<(unsigned)(n / 256), 256, 0, stream>>>(q16, pos, S, H);
    n = (size_t)B * S * KVH * 32;
    rope_kernel<<<(unsigned)(n / 256), 256, 0, stream>>>(k16, pos, S, KVH);
    n = (size_t)B * S * KVH * 64;
    transpose_v_kernel<<<(unsigned)(n / 256), 256, 0, stream>>>(v16, vt16, S, KVH);
  }

  // 4) flash attention: one wave per (b, head, 16-row q tile)
  attn_kernel<<<B * H * (S / 16), 32, 0, stream>>>(q16, k16, vt16, mask, ao16,
                                                   B, S, H, KVH);

  // 5) output projection -> f32 d_out
  {
    int N = E, K = H * D;
    int blocks = ((M / 32) * (N / 64)) / 4;
    wmma_gemm_kernel<float><<<blocks, 128, 0, stream>>>(ao16, wo16, out, M, N, K);
  }
}
